// SplineCNN_27977416966691
// MI455X (gfx1250) — compile-verified, matrix operand order checked by reading, not run on hardware
//
#include <hip/hip_runtime.h>
#include <hip/hip_bf16.h>
#include <stdint.h>

// Problem constants (from reference setup_inputs)
#define B_    32
#define N_    512
#define C0_   128
#define F_    64
#define KK_   9
#define KTOT_ (N_ * KK_)   // 4608 = flattened (m,k) contraction dim

// LDS staging for the aggregation GEMM B panel: 64 rows (f) x 256 K bf16,
// rows padded by 16B (528 = 33*16) so consecutive f rows start 4 banks apart.
#define ROWB_  528
#define BUFB_  (F_ * ROWB_)      // 33792 B per buffer
#define NKB_   (KTOT_ / 256)     // 18 K-blocks of 256

typedef __attribute__((ext_vector_type(16))) __bf16 v16bf;
typedef __attribute__((ext_vector_type(8)))  float  v8f;

__device__ __forceinline__ __bf16 f2bf(float f) {
  unsigned u = __builtin_bit_cast(unsigned, f);
  unsigned r = (u + 0x7FFFu + ((u >> 16) & 1u)) >> 16;   // RNE
  unsigned short s = (unsigned short)r;
  return __builtin_bit_cast(__bf16, s);
}
__device__ __forceinline__ float bf2f(__bf16 b) {
  unsigned short s = __builtin_bit_cast(unsigned short, b);
  unsigned u = ((unsigned)s) << 16;
  return __builtin_bit_cast(float, u);
}

// bf16 A/B fragment: lane holds rows (lane&15); K-runs [half*8,+8) and
// [16+half*8,+8)  -> two contiguous 16B loads when K is the fast dimension.
union FragU { v16bf v; uint4 q[2]; };
__device__ __forceinline__ v16bf load_frag(const __bf16* p) {
  FragU u;
  u.q[0] = *(const uint4*)(p);
  u.q[1] = *(const uint4*)(p + 16);
  return u.v;
}

__device__ __forceinline__ v8f wmma_bf16(v16bf a, v16bf b, v8f c) {
  // v_wmma_f32_16x16x32_bf16
  return __builtin_amdgcn_wmma_f32_16x16x32_bf16(false, a, false, b,
                                                 (short)0, c, false, false);
}

// ---------------------------------------------------------------- conversions
__global__ void k_f32_to_bf16(const float* __restrict__ src,
                              __bf16* __restrict__ dst, int n) {
  int i = blockIdx.x * blockDim.x + threadIdx.x;
  if (i < n) dst[i] = f2bf(src[i]);
}

// src f32 [KK][Cin][F]  ->  dst bf16 [KK][F][Cin]   (root uses KK=1)
__global__ void k_transpose_cf(const float* __restrict__ src,
                               __bf16* __restrict__ dst,
                               int KKp, int Cin, int F) {
  int i = blockIdx.x * blockDim.x + threadIdx.x;
  int tot = KKp * Cin * F;
  if (i >= tot) return;
  int f = i % F;
  int c = (i / F) % Cin;
  int k = i / (F * Cin);
  dst[((size_t)k * F + f) * Cin + c] = f2bf(src[i]);
}

// ---------------------------------------------------------------- degree
__global__ void k_degree(const float* __restrict__ adj,
                         float* __restrict__ invdeg) {
  int n = blockIdx.x, b = blockIdx.y;
  const float* row = adj + ((size_t)b * N_ + n) * N_;
  float s = row[threadIdx.x] + row[threadIdx.x + 256];
  __shared__ float red[256];
  red[threadIdx.x] = s;
  __syncthreads();
  for (int st = 128; st > 0; st >>= 1) {
    if ((int)threadIdx.x < st) red[threadIdx.x] += red[threadIdx.x + st];
    __syncthreads();
  }
  if (threadIdx.x == 0) invdeg[b * N_ + n] = 1.0f / fmaxf(1.0f, red[0]);
}

// ---------------------------------------------------------------- basis
// basis[b][n][m*9+k] (bf16, masked by adj) — the GEMM-2 A matrix, K fastest.
__global__ void k_basis(const float* __restrict__ coord,
                        const float* __restrict__ adj,
                        __bf16* __restrict__ basis) {
  int m = blockIdx.x * blockDim.x + threadIdx.x;
  int n = blockIdx.y, b = blockIdx.z;
  float a = adj[((size_t)b * N_ + n) * N_ + m];
  float bb[2][3];
#pragma unroll
  for (int d = 0; d < 2; ++d) {
    float u = (coord[((size_t)b * N_ + m) * 2 + d]
             - coord[((size_t)b * N_ + n) * 2 + d] + 1.0f) * 0.5f;
    float v  = u * 2.0f;                         // * (K-1)
    float i0 = fminf(fmaxf(floorf(v), 0.0f), 1.0f);
    float fr = v - i0;
    int   i  = (int)i0;
    bb[d][0] = bb[d][1] = bb[d][2] = 0.0f;
    bb[d][i]     = 1.0f - fr;
    bb[d][i + 1] = fr;
  }
  __bf16* out = basis + (((size_t)b * N_ + n) * N_ + m) * KK_;
#pragma unroll
  for (int k0 = 0; k0 < 3; ++k0)
#pragma unroll
    for (int k1 = 0; k1 < 3; ++k1)
      out[k0 * 3 + k1] = f2bf(bb[0][k0] * bb[1][k1] * a);
}

// ---------------------------------------------------------------- GEMM 1
// hwT[b][f][m*9+k] = sum_c h[b][m][c] * w[k][c][f]   (stored transposed so
// GEMM-2 B-fragments are contiguous-K loads).  CIN templated -> full unroll.
template <int CIN>
__global__ void __launch_bounds__(256)
k_hw_gemm(const __bf16* __restrict__ h, const __bf16* __restrict__ wT,
          __bf16* __restrict__ hwT) {
  int b    = blockIdx.y;
  int wid  = threadIdx.x >> 5;
  int lane = threadIdx.x & 31;
  int l15  = lane & 15, half = lane >> 4;
  int id   = blockIdx.x * 8 + wid;       // 0..287 = 9 k-slices x 32 m-tiles
  int k    = id >> 5;
  int m0   = (id & 31) << 4;

  const __bf16* hrow = h + ((size_t)b * N_ + m0 + l15) * CIN + half * 8;
  const __bf16* wrow = wT + ((size_t)k * F_ + l15) * CIN + half * 8;
  v8f zero = {0.f,0.f,0.f,0.f,0.f,0.f,0.f,0.f};
  v8f acc[4] = {zero, zero, zero, zero};

#pragma unroll
  for (int cc = 0; cc < (CIN >> 5); ++cc) {
    v16bf aF = load_frag(hrow + cc * 32);
    v16bf bF[4];
#pragma unroll
    for (int t = 0; t < 4; ++t)
      bF[t] = load_frag(wrow + (size_t)(t * 16) * CIN + cc * 32);
#pragma unroll
    for (int t = 0; t < 4; ++t)
      acc[t] = wmma_bf16(aF, bF[t], acc[t]);
  }
  // C layout: row m = m0 + r + 8*half, col f = t*16 + l15; scatter transposed.
#pragma unroll
  for (int t = 0; t < 4; ++t) {
    int f = t * 16 + l15;
#pragma unroll
    for (int r = 0; r < 8; ++r) {
      int m = m0 + r + half * 8;
      hwT[((size_t)b * F_ + f) * KTOT_ + (size_t)m * KK_ + k] = f2bf(acc[t][r]);
    }
  }
}

// ---------------------------------------------------------------- GEMM 2
// h_out = relu( (basis x hw)/deg + h_in @ root + bias )    fused epilogue
//
// B panel (hwT, shared by all 8 waves of the block) is staged in LDS via
// gfx1250 async loads (global_load_async_to_lds_b128, tracked by ASYNCcnt),
// double-buffered.  A (basis) has zero reuse -> streamed straight to VGPRs.
__device__ __forceinline__ void async_copy_b(const __bf16* __restrict__ hwTb,
                                             unsigned ldsbase, int kb, int tid) {
#pragma unroll
  for (int i = 0; i < 8; ++i) {
    int c   = tid + i * 256;          // 2048 16B-chunks: 64 rows x 32
    int f   = c >> 5;
    int col = c & 31;
    unsigned lds = ldsbase + (unsigned)(f * ROWB_ + col * 16);
    const __bf16* g = hwTb + (size_t)f * KTOT_ + kb * 256 + col * 8;
    asm volatile("global_load_async_to_lds_b128 %0, %1, off"
                 :: "v"(lds), "v"(g) : "memory");
  }
}

template <int CIN>
__global__ void __launch_bounds__(256)
k_agg_gemm(const __bf16* __restrict__ basis, const __bf16* __restrict__ hwT,
           const __bf16* __restrict__ h_in,  const __bf16* __restrict__ rootT,
           const float* __restrict__ bias,   const float* __restrict__ invdeg,
           __bf16* __restrict__ h_out) {
  __shared__ char smem[2 * BUFB_];

  int b    = blockIdx.y;
  int tid  = threadIdx.x;
  int wid  = tid >> 5;
  int lane = tid & 31;
  int l15  = lane & 15, half = lane >> 4;
  int n0   = blockIdx.x * 128 + wid * 16;

  const __bf16* arow = basis + ((size_t)b * N_ + n0 + l15) * (size_t)KTOT_
                             + half * 8;
  const __bf16* hwTb = hwT + (size_t)b * F_ * KTOT_;
  unsigned smembase = (unsigned)(uintptr_t)(void*)&smem[0];

  v8f zero = {0.f,0.f,0.f,0.f,0.f,0.f,0.f,0.f};
  v8f acc[4] = {zero, zero, zero, zero};

  async_copy_b(hwTb, smembase, 0, tid);               // prime buffer 0

  for (int kb = 0; kb < NKB_; ++kb) {
    int cur = kb & 1;
    if (kb + 1 < NKB_) {
      async_copy_b(hwTb, smembase + (cur ^ 1) * BUFB_, kb + 1, tid);
      asm volatile("s_wait_asynccnt 0x8" ::: "memory");  // cur buffer complete
    } else {
      asm volatile("s_wait_asynccnt 0x0" ::: "memory");
    }
    __syncthreads();                                   // all waves' parts in LDS

    // preload this K-block's 8 A fragments (batched global_load_b128 pairs)
    v16bf aF[8];
#pragma unroll
    for (int j = 0; j < 8; ++j) {
      aF[j] = load_frag(arow + (size_t)(kb * 8 + j) * 32);
      __builtin_prefetch(arow + (size_t)(kb * 8 + j) * 32 + 512, 0, 0);
    }
    const char* bbase = smem + cur * BUFB_ + half * 16;
#pragma unroll
    for (int j = 0; j < 8; ++j) {
#pragma unroll
      for (int t = 0; t < 4; ++t) {
        const __bf16* bp =
            (const __bf16*)(bbase + (t * 16 + l15) * ROWB_ + j * 64);
        acc[t] = wmma_bf16(aF[j], load_frag(bp), acc[t]);
      }
    }
    __syncthreads();   // done reading buf[cur] before it is refilled
  }

  // per-row 1/deg scale: accumulator row = n0 + r + 8*half (C-layout)
#pragma unroll
  for (int r = 0; r < 8; ++r) {
    float s = invdeg[b * N_ + n0 + r + half * 8];
#pragma unroll
    for (int t = 0; t < 4; ++t) acc[t][r] *= s;
  }

  // + h_in @ root (short K loop over CIN, fully unrolled)
  const __bf16* hrow = h_in + ((size_t)b * N_ + n0 + l15) * CIN + half * 8;
  const __bf16* rrow = rootT + (size_t)l15 * CIN + half * 8;
#pragma unroll
  for (int cc = 0; cc < (CIN >> 5); ++cc) {
    v16bf aF = load_frag(hrow + cc * 32);
    v16bf bF[4];
#pragma unroll
    for (int t = 0; t < 4; ++t)
      bF[t] = load_frag(rrow + (size_t)(t * 16) * CIN + cc * 32);
#pragma unroll
    for (int t = 0; t < 4; ++t)
      acc[t] = wmma_bf16(aF, bF[t], acc[t]);
  }

  // bias + relu + bf16 store
#pragma unroll
  for (int t = 0; t < 4; ++t) {
    float bia = bias[t * 16 + l15];
#pragma unroll
    for (int r = 0; r < 8; ++r) {
      float v = acc[t][r] + bia;
      v = v > 0.f ? v : 0.f;
      h_out[((size_t)b * N_ + n0 + r + half * 8) * F_ + t * 16 + l15] = f2bf(v);
    }
  }
}

// ---------------------------------------------------------------- pool + FC
__global__ void k_pool_fc(const __bf16* __restrict__ h3,
                          const float* __restrict__ fc_w,
                          const float* __restrict__ fc_b,
                          float* __restrict__ out) {
  int b = blockIdx.x;
  int f = threadIdx.x;              // 0..63
  float m = -3.4e38f;
  for (int n = 0; n < N_; ++n)
    m = fmaxf(m, bf2f(h3[((size_t)b * N_ + n) * F_ + f]));
  __shared__ float g[F_];
  g[f] = m;
  __syncthreads();
  if (f < 10) {
    float o = fc_b[f];
    for (int c = 0; c < F_; ++c) o += g[c] * fc_w[c * 10 + f];
    out[b * 10 + f] = o;
  }
}

// ---------------------------------------------------------------- launch
extern "C" void kernel_launch(void* const* d_in, const int* in_sizes, int n_in,
                              void* d_out, int out_size, void* d_ws, size_t ws_size,
                              hipStream_t stream) {
  (void)in_sizes; (void)n_in; (void)out_size; (void)ws_size;
  const float* x     = (const float*)d_in[0];
  const float* coord = (const float*)d_in[1];
  const float* adj   = (const float*)d_in[2];
  const float* w[3]    = {(const float*)d_in[3], (const float*)d_in[6], (const float*)d_in[9]};
  const float* root[3] = {(const float*)d_in[4], (const float*)d_in[7], (const float*)d_in[10]};
  const float* bias[3] = {(const float*)d_in[5], (const float*)d_in[8], (const float*)d_in[11]};
  const float* fc_w = (const float*)d_in[12];
  const float* fc_b = (const float*)d_in[13];
  float* out = (float*)d_out;

  // workspace carve-up (256B aligned)
  char* ws = (char*)d_ws;
  size_t off = 0;
  auto alloc = [&](size_t bytes) {
    size_t o = off; off += (bytes + 255) & ~(size_t)255; return o;
  };
  __bf16* basis = (__bf16*)(ws + alloc((size_t)B_ * N_ * N_ * KK_ * 2)); // 151 MB
  __bf16* hwT   = (__bf16*)(ws + alloc((size_t)B_ * F_ * KTOT_ * 2));   // 18.9 MB
  __bf16* h[4];
  h[0] = (__bf16*)(ws + alloc((size_t)B_ * N_ * C0_ * 2));
  for (int l = 1; l < 4; ++l) h[l] = (__bf16*)(ws + alloc((size_t)B_ * N_ * F_ * 2));
  float* invdeg = (float*)(ws + alloc((size_t)B_ * N_ * 4));
  int cin[3] = {C0_, F_, F_};
  __bf16* wT[3]; __bf16* rootT[3];
  for (int l = 0; l < 3; ++l) {
    wT[l]    = (__bf16*)(ws + alloc((size_t)KK_ * F_ * cin[l] * 2));
    rootT[l] = (__bf16*)(ws + alloc((size_t)F_ * cin[l] * 2));
  }

  // weight / input conversions
  for (int l = 0; l < 3; ++l) {
    int totw = KK_ * cin[l] * F_;
    k_transpose_cf<<<(totw + 255) / 256, 256, 0, stream>>>(w[l], wT[l], KK_, cin[l], F_);
    int totr = cin[l] * F_;
    k_transpose_cf<<<(totr + 255) / 256, 256, 0, stream>>>(root[l], rootT[l], 1, cin[l], F_);
  }
  int totx = B_ * N_ * C0_;
  k_f32_to_bf16<<<(totx + 255) / 256, 256, 0, stream>>>(x, h[0], totx);

  // degree + masked spline basis (written once, reused by all 3 layers)
  k_degree<<<dim3(N_, B_), 256, 0, stream>>>(adj, invdeg);
  k_basis<<<dim3(N_ / 256, N_, B_), 256, 0, stream>>>(coord, adj, basis);

  // layer 0 (Cin=128), layers 1-2 (Cin=64)
  k_hw_gemm<C0_><<<dim3(36, B_), 256, 0, stream>>>(h[0], wT[0], hwT);
  k_agg_gemm<C0_><<<dim3(N_ / 128, B_), 256, 0, stream>>>(basis, hwT, h[0], rootT[0],
                                                          bias[0], invdeg, h[1]);
  for (int l = 1; l < 3; ++l) {
    k_hw_gemm<F_><<<dim3(36, B_), 256, 0, stream>>>(h[l], wT[l], hwT);
    k_agg_gemm<F_><<<dim3(N_ / 128, B_), 256, 0, stream>>>(basis, hwT, h[l], rootT[l],
                                                           bias[l], invdeg, h[l + 1]);
  }

  // global max-pool over nodes + final FC
  k_pool_fc<<<B_, F_, 0, stream>>>(h[3], fc_w, fc_b, out);
}
